// RNN_SQ2SQ_42305427865838
// MI455X (gfx1250) — compile-verified, matrix operand order checked by reading, not run on hardware
//
#include <hip/hip_runtime.h>
#include <hip/hip_bf16.h>

// ---------------- model dims ----------------
#define BB   64     // batch
#define LINN 256    // encoder sequence length
#define INF  128    // input features
#define HH   512    // hidden
#define H3   1536   // 3*H
#define HOR  64     // decoder horizon
#define ROWS (BB*LINN)  // 16384

#define RNN_BLOCKS 16   // persistent kernels: 16 blocks x 8 waves = 128 waves
                        // == number of 16x16 column-triple jobs per phase

// ---------------- types / helpers ----------------
typedef __attribute__((ext_vector_type(8)))  __bf16 bf16x8;
typedef __attribute__((ext_vector_type(16))) __bf16 v16bf;
typedef __attribute__((ext_vector_type(8)))  float  v8f;

#define DEVINL __device__ __forceinline__

// Load a 16x32 bf16 tile fragment (A row-major, or B stored column-major with
// contiguous K) into the wave32 WMMA register layout (ISA 7.12.2):
// lane = m + 16*half ; element i holds k = (i<8 ? i : i+8) + 8*half
DEVINL v16bf load_frag(const __bf16* __restrict__ base, int ld, int lane) {
  const int m = lane & 15, hf = lane >> 4;
  const __bf16* p = base + (size_t)m * ld + (hf << 3);
  bf16x8 lo = *(const bf16x8*)(p);
  bf16x8 hi = *(const bf16x8*)(p + 16);
  return __builtin_shufflevector(lo, hi, 0,1,2,3,4,5,6,7,8,9,10,11,12,13,14,15);
}

DEVINL v8f wmma_bf16(v16bf a, v16bf b, v8f c) {
  return __builtin_amdgcn_wmma_f32_16x16x32_bf16(false, a, false, b, (short)0, c,
                                                 false, false);
}

DEVINL float sig_f(float x)  { return 1.f / (1.f + __expf(-x)); }
DEVINL float tanh_f(float x) { return 2.f / (1.f + __expf(-2.f * x)) - 1.f; }

// Monotonic-counter grid barrier (persistent kernels, co-resident blocks).
DEVINL void grid_barrier(unsigned* cnt, unsigned target) {
  __syncthreads();
  if (threadIdx.x == 0) {
    __hip_atomic_fetch_add(cnt, 1u, __ATOMIC_RELEASE, __HIP_MEMORY_SCOPE_AGENT);
    while (__hip_atomic_load(cnt, __ATOMIC_ACQUIRE, __HIP_MEMORY_SCOPE_AGENT) < target)
      __builtin_amdgcn_s_sleep(2);
  }
  __syncthreads();
}

// ---------------- prep kernels ----------------
// Wt[n*K + k] = (bf16) W[k*N + n]   (weights become column-major / K-contiguous)
__global__ void k_transpose_bf16(const float* __restrict__ W, __bf16* __restrict__ Wt,
                                 int K, int N) {
  const int total = K * N;
  for (int id = blockIdx.x * blockDim.x + threadIdx.x; id < total;
       id += gridDim.x * blockDim.x) {
    const int n = id / K, k = id - n * K;
    Wt[id] = (__bf16)W[(size_t)k * N + n];
  }
}

__global__ void k_cvt_bf16(const float* __restrict__ X, __bf16* __restrict__ Y, int n) {
  for (int id = blockIdx.x * blockDim.x + threadIdx.x; id < n;
       id += gridDim.x * blockDim.x)
    Y[id] = (__bf16)X[id];
}

// ---------------- generic WMMA GEMM: C(rows,N) = A(rows,K) @ Wt^T + bias ------
// A bf16 row-major (lda=K), Wt bf16 [N][K]. One wave -> 16x64 output strip.
// remap!=0 : input row r = b*LIN + t is stored at output row t*B + b (time-major).
__global__ void k_gemm_bf16(const __bf16* __restrict__ A, int lda,
                            const __bf16* __restrict__ Wt, int K,
                            const float* __restrict__ bias,
                            float* __restrict__ C, int rows, int N, int remap) {
  const int lane = threadIdx.x & 31;
  const int gw = (blockIdx.x * blockDim.x + threadIdx.x) >> 5;
  const int nw = (gridDim.x * blockDim.x) >> 5;
  const int colg = N >> 6;
  const int jobs = (rows >> 4) * colg;
  for (int job = gw; job < jobs; job += nw) {
    const int rt = job / colg, cg = job - rt * colg;
    const int r0 = rt << 4, n0 = cg << 6;
    v8f a0 = {}, a1 = {}, a2 = {}, a3 = {};
    for (int k0 = 0; k0 < K; k0 += 32) {
      v16bf a  = load_frag(A  + (size_t)r0 * lda + k0, lda, lane);
      v16bf b0 = load_frag(Wt + (size_t)(n0     ) * K + k0, K, lane);
      v16bf b1 = load_frag(Wt + (size_t)(n0 + 16) * K + k0, K, lane);
      v16bf b2 = load_frag(Wt + (size_t)(n0 + 32) * K + k0, K, lane);
      v16bf b3 = load_frag(Wt + (size_t)(n0 + 48) * K + k0, K, lane);
      a0 = wmma_bf16(a, b0, a0); a1 = wmma_bf16(a, b1, a1);
      a2 = wmma_bf16(a, b2, a2); a3 = wmma_bf16(a, b3, a3);
    }
    const int n = lane & 15, hf = lane >> 4;
    const float bb0 = bias ? bias[n0      + n] : 0.f;
    const float bb1 = bias ? bias[n0 + 16 + n] : 0.f;
    const float bb2 = bias ? bias[n0 + 32 + n] : 0.f;
    const float bb3 = bias ? bias[n0 + 48 + n] : 0.f;
    for (int r = 0; r < 8; ++r) {
      const int row = r0 + r + (hf << 3);
      const int orow = remap ? (((row & (LINN - 1)) << 6) | (row >> 8)) : row;
      float* crow = C + (size_t)orow * N + n0 + n;
      crow[0]  = a0[r] + bb0;  crow[16] = a1[r] + bb1;
      crow[32] = a2[r] + bb2;  crow[48] = a3[r] + bb3;
    }
  }
}

// ---------------- encoder GRU recurrence (persistent) ---------------
// gi: precomputed x@Wih+bih, time-major [t*B + b][3H].  Per step:
// gh = h@Whh, gates fused on WMMA accumulators, ping-pong h, one barrier/step.
__global__ void k_gru_enc(const float* __restrict__ gi,
                          const __bf16* __restrict__ Whht,   // [3H][H]
                          const float* __restrict__ bhh,
                          float* __restrict__ hf0, float* __restrict__ hf1,
                          __bf16* __restrict__ hb0, __bf16* __restrict__ hb1,
                          __bf16* __restrict__ out_bf,       // [LIN*B][H]
                          float*  __restrict__ out_f,        // optional
                          unsigned* __restrict__ cnt) {
  const int lane = threadIdx.x & 31;
  const int gw = (blockIdx.x * blockDim.x + threadIdx.x) >> 5;
  const int nw = (gridDim.x * blockDim.x) >> 5;
  unsigned epoch = 0;
  for (int t = 0; t < LINN; ++t) {
    const float*  hsf = (t & 1) ? hf1 : hf0;
    const __bf16* hsb = (t & 1) ? hb1 : hb0;
    float*  hdf = (t & 1) ? hf0 : hf1;
    __bf16* hdb = (t & 1) ? hb0 : hb1;
    const float* git = gi + (size_t)t * BB * H3;
    // 4 row-tiles x 32 column-triples (r/z/n gates share a triple) = 128 jobs
    for (int tr = gw; tr < 128; tr += nw) {
      const int r0 = (tr & 3) << 4, j0 = (tr >> 2) << 4;
      v8f aR = {}, aZ = {}, aN = {};
      for (int k0 = 0; k0 < HH; k0 += 32) {
        v16bf a  = load_frag(hsb + (size_t)r0 * HH + k0, HH, lane);
        v16bf bR = load_frag(Whht + (size_t)(j0          ) * HH + k0, HH, lane);
        v16bf bZ = load_frag(Whht + (size_t)(j0 + HH     ) * HH + k0, HH, lane);
        v16bf bN = load_frag(Whht + (size_t)(j0 + 2 * HH ) * HH + k0, HH, lane);
        aR = wmma_bf16(a, bR, aR); aZ = wmma_bf16(a, bZ, aZ); aN = wmma_bf16(a, bN, aN);
      }
      const int n = lane & 15, hf = lane >> 4;
      const int j = j0 + n;
      const float bR_ = bhh[j], bZ_ = bhh[HH + j], bN_ = bhh[2 * HH + j];
      for (int r = 0; r < 8; ++r) {
        const int m = r0 + r + (hf << 3);                       // batch index
        const float* gir = git + (size_t)m * H3 + j;
        const float rr = sig_f(gir[0]      + aR[r] + bR_);
        const float zz = sig_f(gir[HH]     + aZ[r] + bZ_);
        const float nn = tanh_f(gir[2 * HH] + rr * (aN[r] + bN_));
        const float hn = (1.f - zz) * nn + zz * hsf[m * HH + j];
        hdf[m * HH + j] = hn;
        hdb[m * HH + j] = (__bf16)hn;
        const size_t o = ((size_t)t * BB + m) * HH + j;
        out_bf[o] = (__bf16)hn;
        if (out_f) out_f[o] = hn;
      }
    }
    ++epoch;
    grid_barrier(cnt, epoch * gridDim.x);
  }
}

// ---------------- decoder ----------------
struct DecArgs {
  const float*  src;        // (B, LIN, IN) f32
  const float*  encproj;    // [LIN*B][H] f32   (enc_out @ Wa_enc)
  const float*  encout_f;   // [LIN*B][H] f32
  const __bf16* Wat_h;      // [H][2H]
  const float*  ba; const float* va;
  const __bf16* Wih1t;      // [3H][IN+H]
  const __bf16* Whh1t;      // [3H][H]
  const float*  bih1; const float* bhh1;
  const __bf16* Wih2t;      // [3H][H]
  const __bf16* Whh2t;      // [3H][H]
  const float*  bih2; const float* bhh2;
  const __bf16* Wfct;       // [IN][H+H+IN]
  const float*  bfc;
  float* h1f[2]; __bf16* h1b[2];
  float* h2f[2]; __bf16* h2b[2];
  float*  hproj;            // [B][H]
  float*  scores;           // [B][LIN]
  __bf16* wbf;              // [B][H]
  __bf16* inpb[2];          // [B][IN]
  float*  out;              // (B, HOR, IN)
  unsigned* cnt;
};

// one GRU layer with fused gates; gi A-source split into up to 2 K-segments
DEVINL void dec_gru(int lane, int gw, int nw,
                    const __bf16* xA0, int ld0, int K0,
                    const __bf16* xA1, int ld1, int K1,
                    const __bf16* Wiht, int ldwi,
                    const __bf16* hb_src, const float* hf_src,
                    const __bf16* Whht,
                    const float* bih, const float* bhh,
                    float* hf_dst, __bf16* hb_dst) {
  for (int tr = gw; tr < 128; tr += nw) {
    const int r0 = (tr & 3) << 4, j0 = (tr >> 2) << 4;
    v8f giR = {}, giZ = {}, giN = {}, ghR = {}, ghZ = {}, ghN = {};
    for (int s = 0; s < 2; ++s) {
      const __bf16* Ap = s ? xA1 : xA0;
      const int ld = s ? ld1 : ld0, Ks = s ? K1 : K0, kb = s ? K0 : 0;
      if (!Ap || !Ks) continue;
      for (int kl = 0; kl < Ks; kl += 32) {
        v16bf a  = load_frag(Ap + (size_t)r0 * ld + kl, ld, lane);
        const int kg = kb + kl;
        v16bf bR = load_frag(Wiht + (size_t)(j0          ) * ldwi + kg, ldwi, lane);
        v16bf bZ = load_frag(Wiht + (size_t)(j0 + HH     ) * ldwi + kg, ldwi, lane);
        v16bf bN = load_frag(Wiht + (size_t)(j0 + 2 * HH ) * ldwi + kg, ldwi, lane);
        giR = wmma_bf16(a, bR, giR); giZ = wmma_bf16(a, bZ, giZ); giN = wmma_bf16(a, bN, giN);
      }
    }
    for (int kl = 0; kl < HH; kl += 32) {
      v16bf a  = load_frag(hb_src + (size_t)r0 * HH + kl, HH, lane);
      v16bf bR = load_frag(Whht + (size_t)(j0          ) * HH + kl, HH, lane);
      v16bf bZ = load_frag(Whht + (size_t)(j0 + HH     ) * HH + kl, HH, lane);
      v16bf bN = load_frag(Whht + (size_t)(j0 + 2 * HH ) * HH + kl, HH, lane);
      ghR = wmma_bf16(a, bR, ghR); ghZ = wmma_bf16(a, bZ, ghZ); ghN = wmma_bf16(a, bN, ghN);
    }
    const int n = lane & 15, hf = lane >> 4;
    const int j = j0 + n;
    const float biR = bih[j], biZ = bih[HH + j], biN = bih[2 * HH + j];
    const float bhR = bhh[j], bhZ = bhh[HH + j], bhN = bhh[2 * HH + j];
    for (int r = 0; r < 8; ++r) {
      const int m = r0 + r + (hf << 3);
      const float rr = sig_f(giR[r] + biR + ghR[r] + bhR);
      const float zz = sig_f(giZ[r] + biZ + ghZ[r] + bhZ);
      const float nn = tanh_f(giN[r] + biN + rr * (ghN[r] + bhN));
      const float hn = (1.f - zz) * nn + zz * hf_src[m * HH + j];
      hf_dst[m * HH + j] = hn;
      hb_dst[m * HH + j] = (__bf16)hn;
    }
  }
}

__global__ void k_decoder(DecArgs A) {
  const int tid  = threadIdx.x;
  const int lane = tid & 31;
  const int gtid = blockIdx.x * blockDim.x + tid;
  const int nth  = gridDim.x * blockDim.x;
  const int gw   = gtid >> 5;
  const int nw   = nth >> 5;
  unsigned epoch = 0;

  // init: inp[0] = source[:, -1, :]
  for (int i = gtid; i < BB * INF; i += nth) {
    const int b = i >> 7, c = i & (INF - 1);
    A.inpb[0][i] = (__bf16)A.src[((size_t)b * LINN + (LINN - 1)) * INF + c];
  }
  ++epoch; grid_barrier(A.cnt, epoch * gridDim.x);

  for (int d = 0; d < HOR; ++d) {
    const int cur = d & 1, nxt = cur ^ 1;

    // P1: hproj = [h1,h2] @ Wat_h + ba       (4x32 tiles, segmented K)
    for (int tile = gw; tile < 128; tile += nw) {
      const int r0 = (tile & 3) << 4, n0 = (tile >> 2) << 4;
      v8f acc = {};
      for (int s = 0; s < 2; ++s) {
        const __bf16* hs = s ? A.h2b[cur] : A.h1b[cur];
        for (int kl = 0; kl < HH; kl += 32) {
          v16bf a = load_frag(hs + (size_t)r0 * HH + kl, HH, lane);
          v16bf b = load_frag(A.Wat_h + (size_t)n0 * (2 * HH) + s * HH + kl, 2 * HH, lane);
          acc = wmma_bf16(a, b, acc);
        }
      }
      const int n = lane & 15, hf = lane >> 4;
      const float bav = A.ba[n0 + n];
      for (int r = 0; r < 8; ++r)
        A.hproj[(r0 + r + (hf << 3)) * HH + n0 + n] = acc[r] + bav;
    }
    ++epoch; grid_barrier(A.cnt, epoch * gridDim.x);

    // P2: scores[b][l] = va . tanh(encproj[l,b,:] + hproj[b,:])
    for (int idx = gtid; idx < BB * LINN; idx += nth) {
      const int b = idx & (BB - 1), l = idx >> 6;
      const float* ep = A.encproj + ((size_t)l * BB + b) * HH;
      const float* hp = A.hproj + (size_t)b * HH;
      float s = 0.f;
      for (int j = 0; j < HH; ++j) s += A.va[j] * tanh_f(ep[j] + hp[j]);
      A.scores[b * LINN + l] = s;
    }
    ++epoch; grid_barrier(A.cnt, epoch * gridDim.x);

    // P3: softmax over l, one wave per batch row
    for (int b = gw; b < BB; b += nw) {
      float v[8]; float mx = -1e30f;
      for (int i = 0; i < 8; ++i) {
        v[i] = A.scores[b * LINN + lane + 32 * i];
        mx = fmaxf(mx, v[i]);
      }
      for (int off = 16; off; off >>= 1) mx = fmaxf(mx, __shfl_xor(mx, off, 32));
      float sum = 0.f;
      for (int i = 0; i < 8; ++i) { v[i] = __expf(v[i] - mx); sum += v[i]; }
      for (int off = 16; off; off >>= 1) sum += __shfl_xor(sum, off, 32);
      const float inv = 1.f / sum;
      for (int i = 0; i < 8; ++i) A.scores[b * LINN + lane + 32 * i] = v[i] * inv;
    }
    ++epoch; grid_barrier(A.cnt, epoch * gridDim.x);

    // P4: w[b] = sum_l a[b,l] * enc_out[l,b,:]
    // 128 jobs: (b, column-half); 8 contiguous cols per lane.
    for (int job = gw; job < BB * 2; job += nw) {
      const int b = job >> 1;
      const int c0 = ((job & 1) << 8) + lane * 8;
      float acc[8];
      for (int i = 0; i < 8; ++i) acc[i] = 0.f;
      for (int l = 0; l < LINN; ++l) {
        const float av = A.scores[b * LINN + l];
        const float* eo = A.encout_f + ((size_t)l * BB + b) * HH + c0;
        for (int i = 0; i < 8; ++i) acc[i] += av * eo[i];
      }
      for (int i = 0; i < 8; ++i)
        A.wbf[b * HH + c0 + i] = (__bf16)acc[i];
    }
    ++epoch; grid_barrier(A.cnt, epoch * gridDim.x);

    // P5: GRU layer 1, x = [inp | w]
    dec_gru(lane, gw, nw,
            A.inpb[cur], INF, INF, A.wbf, HH, HH,
            A.Wih1t, INF + HH,
            A.h1b[cur], A.h1f[cur], A.Whh1t,
            A.bih1, A.bhh1, A.h1f[nxt], A.h1b[nxt]);
    ++epoch; grid_barrier(A.cnt, epoch * gridDim.x);

    // P6: GRU layer 2, x = h1_new
    dec_gru(lane, gw, nw,
            A.h1b[nxt], HH, HH, (const __bf16*)nullptr, 0, 0,
            A.Wih2t, HH,
            A.h2b[cur], A.h2f[cur], A.Whh2t,
            A.bih2, A.bhh2, A.h2f[nxt], A.h2b[nxt]);
    ++epoch; grid_barrier(A.cnt, epoch * gridDim.x);

    // P7: pred = [h2_new | w | inp] @ Wfc + bfc   (4x8 tiles, 3 K-segments)
    for (int tile = gw; tile < 32; tile += nw) {
      const int r0 = (tile & 3) << 4, n0 = (tile >> 2) << 4;
      const int ldw = HH + HH + INF;
      v8f acc = {};
      for (int kl = 0; kl < HH; kl += 32) {
        v16bf a = load_frag(A.h2b[nxt] + (size_t)r0 * HH + kl, HH, lane);
        v16bf b = load_frag(A.Wfct + (size_t)n0 * ldw + kl, ldw, lane);
        acc = wmma_bf16(a, b, acc);
      }
      for (int kl = 0; kl < HH; kl += 32) {
        v16bf a = load_frag(A.wbf + (size_t)r0 * HH + kl, HH, lane);
        v16bf b = load_frag(A.Wfct + (size_t)n0 * ldw + HH + kl, ldw, lane);
        acc = wmma_bf16(a, b, acc);
      }
      for (int kl = 0; kl < INF; kl += 32) {
        v16bf a = load_frag(A.inpb[cur] + (size_t)r0 * INF + kl, INF, lane);
        v16bf b = load_frag(A.Wfct + (size_t)n0 * ldw + 2 * HH + kl, ldw, lane);
        acc = wmma_bf16(a, b, acc);
      }
      const int n = lane & 15, hf = lane >> 4;
      const float bv = A.bfc[n0 + n];
      for (int r = 0; r < 8; ++r) {
        const int m = r0 + r + (hf << 3);
        const float pv = acc[r] + bv;
        A.out[((size_t)m * HOR + d) * INF + n0 + n] = pv;
        A.inpb[nxt][m * INF + n0 + n] = (__bf16)pv;
      }
    }
    ++epoch; grid_barrier(A.cnt, epoch * gridDim.x);
  }
}

// ---------------- host ----------------
extern "C" void kernel_launch(void* const* d_in, const int* in_sizes, int n_in,
                              void* d_out, int out_size, void* d_ws, size_t ws_size,
                              hipStream_t stream) {
  (void)in_sizes; (void)n_in; (void)out_size; (void)ws_size;
  const float* src   = (const float*)d_in[0];
  const float* eWih1 = (const float*)d_in[1];
  const float* eWhh1 = (const float*)d_in[2];
  const float* ebih1 = (const float*)d_in[3];
  const float* ebhh1 = (const float*)d_in[4];
  const float* eWih2 = (const float*)d_in[5];
  const float* eWhh2 = (const float*)d_in[6];
  const float* ebih2 = (const float*)d_in[7];
  const float* ebhh2 = (const float*)d_in[8];
  const float* dWih1 = (const float*)d_in[9];
  const float* dWhh1 = (const float*)d_in[10];
  const float* dbih1 = (const float*)d_in[11];
  const float* dbhh1 = (const float*)d_in[12];
  const float* dWih2 = (const float*)d_in[13];
  const float* dWhh2 = (const float*)d_in[14];
  const float* dbih2 = (const float*)d_in[15];
  const float* dbhh2 = (const float*)d_in[16];
  const float* Wa    = (const float*)d_in[17];
  const float* ba    = (const float*)d_in[18];
  const float* va    = (const float*)d_in[19];
  const float* Wfc   = (const float*)d_in[20];
  const float* bfc   = (const float*)d_in[21];

  char* p = (char*)d_ws;
  auto take = [&](size_t bytes) -> char* {
    char* r = p;
    p += (bytes + 255) & ~(size_t)255;
    return r;
  };

  // ---- small (zeroed) region: counters + decoder/recurrence state ----
  char* small0 = p;
  unsigned* cnt = (unsigned*)take(256);
  float*  h1f0 = (float*)take(BB * HH * 4);
  float*  h1f1 = (float*)take(BB * HH * 4);
  float*  h2f0 = (float*)take(BB * HH * 4);
  float*  h2f1 = (float*)take(BB * HH * 4);
  __bf16* h1b0 = (__bf16*)take(BB * HH * 2);
  __bf16* h1b1 = (__bf16*)take(BB * HH * 2);
  __bf16* h2b0 = (__bf16*)take(BB * HH * 2);
  __bf16* h2b1 = (__bf16*)take(BB * HH * 2);
  float*  hproj  = (float*)take(BB * HH * 4);
  float*  scores = (float*)take(BB * LINN * 4);
  __bf16* wbf    = (__bf16*)take(BB * HH * 2);
  __bf16* inpb0  = (__bf16*)take(BB * INF * 2);
  __bf16* inpb1  = (__bf16*)take(BB * INF * 2);
  const size_t small_bytes = (size_t)(p - small0);

  // ---- bf16 transposed weights ----
  __bf16* Wih1t  = (__bf16*)take((size_t)H3 * INF * 2);
  __bf16* Whh1t  = (__bf16*)take((size_t)H3 * HH * 2);
  __bf16* Wih2t  = (__bf16*)take((size_t)H3 * HH * 2);
  __bf16* Whh2t  = (__bf16*)take((size_t)H3 * HH * 2);
  __bf16* DWih1t = (__bf16*)take((size_t)H3 * (INF + HH) * 2);
  __bf16* DWhh1t = (__bf16*)take((size_t)H3 * HH * 2);
  __bf16* DWih2t = (__bf16*)take((size_t)H3 * HH * 2);
  __bf16* DWhh2t = (__bf16*)take((size_t)H3 * HH * 2);
  __bf16* Wat_h  = (__bf16*)take((size_t)HH * (2 * HH) * 2);
  __bf16* Wat_e  = (__bf16*)take((size_t)HH * HH * 2);
  __bf16* Wfct   = (__bf16*)take((size_t)INF * (2 * HH + INF) * 2);

  // ---- big buffers ----
  float*  gi      = (float*)take((size_t)ROWS * H3 * 4);
  __bf16* enc1b   = (__bf16*)take((size_t)ROWS * HH * 2);
  __bf16* enc2b   = (__bf16*)take((size_t)ROWS * HH * 2);
  float*  enc2f   = (float*)take((size_t)ROWS * HH * 4);
  float*  encproj = (float*)take((size_t)ROWS * HH * 4);
  __bf16* srcb    = (__bf16*)take((size_t)ROWS * INF * 2);

  hipMemsetAsync(small0, 0, small_bytes, stream);

  // weight prep
  auto tl = [&](const float* W, __bf16* Wt, int K, int N) {
    const int total = K * N;
    int blocks = (total + 255) / 256;
    if (blocks > 4096) blocks = 4096;
    k_transpose_bf16<<<blocks, 256, 0, stream>>>(W, Wt, K, N);
  };
  tl(eWih1, Wih1t, INF, H3);
  tl(eWhh1, Whh1t, HH, H3);
  tl(eWih2, Wih2t, HH, H3);
  tl(eWhh2, Whh2t, HH, H3);
  tl(dWih1, DWih1t, INF + HH, H3);
  tl(dWhh1, DWhh1t, HH, H3);
  tl(dWih2, DWih2t, HH, H3);
  tl(dWhh2, DWhh2t, HH, H3);
  tl(Wa, Wat_h, 2 * HH, HH);                       // Wa rows 0..1023 (h part)
  tl(Wa + (size_t)(2 * HH) * HH, Wat_e, HH, HH);   // Wa rows 1024..1535 (enc part)
  tl(Wfc, Wfct, 2 * HH + INF, INF);
  k_cvt_bf16<<<2048, 256, 0, stream>>>(src, srcb, ROWS * INF);

  // encoder: hoisted input GEMMs + persistent recurrences
  k_gemm_bf16<<<1024, 256, 0, stream>>>(srcb, INF, Wih1t, INF, ebih1,
                                        gi, ROWS, H3, 1 /*remap to time-major*/);
  k_gru_enc<<<RNN_BLOCKS, 256, 0, stream>>>(gi, Whh1t, ebhh1, h1f0, h1f1, h1b0, h1b1,
                                            enc1b, nullptr, cnt + 0);
  k_gemm_bf16<<<1024, 256, 0, stream>>>(enc1b, HH, Wih2t, HH, ebih2,
                                        gi, ROWS, H3, 0);
  k_gru_enc<<<RNN_BLOCKS, 256, 0, stream>>>(gi, Whh2t, ebhh2, h2f0, h2f1, h2b0, h2b1,
                                            enc2b, enc2f, cnt + 16);

  // step-invariant attention projection: enc_out @ Wa_enc
  k_gemm_bf16<<<1024, 256, 0, stream>>>(enc2b, HH, Wat_e, HH, nullptr,
                                        encproj, ROWS, HH, 0);

  // decoder (persistent)
  DecArgs da;
  da.src = src; da.encproj = encproj; da.encout_f = enc2f;
  da.Wat_h = Wat_h; da.ba = ba; da.va = va;
  da.Wih1t = DWih1t; da.Whh1t = DWhh1t; da.bih1 = dbih1; da.bhh1 = dbhh1;
  da.Wih2t = DWih2t; da.Whh2t = DWhh2t; da.bih2 = dbih2; da.bhh2 = dbhh2;
  da.Wfct = Wfct; da.bfc = bfc;
  da.h1f[0] = h1f0; da.h1f[1] = h1f1; da.h1b[0] = h1b0; da.h1b[1] = h1b1;
  da.h2f[0] = h2f0; da.h2f[1] = h2f1; da.h2b[0] = h2b0; da.h2b[1] = h2b1;
  da.hproj = hproj; da.scores = scores; da.wbf = wbf;
  da.inpb[0] = inpb0; da.inpb[1] = inpb1;
  da.out = (float*)d_out; da.cnt = cnt + 32;
  k_decoder<<<RNN_BLOCKS, 256, 0, stream>>>(da);
}
